// RandomWalkGAT_6433861009606
// MI455X (gfx1250) — matrix operand order repaired and checked by reading, not compiled, not run on hardware
//
#include <hip/hip_runtime.h>
#include <hip/hip_bf16.h>
#include <cstdint>
#include <cstddef>

// ---------------------------------------------------------------------------
// GAT (2x GATConv) pipeline for MI455X (gfx1250, wave32, WMMA).
// GEMMs use v_wmma_f32_16x16x32_f16; edge softmax/aggregation use native
// float atomics (global_atomic_add_f32 + int-trick atomic max).
// ---------------------------------------------------------------------------

typedef _Float16 half_t;
typedef __attribute__((ext_vector_type(8)))  _Float16 v8h;
typedef __attribute__((ext_vector_type(16))) _Float16 v16h;
typedef __attribute__((ext_vector_type(8)))  float    v8f;

#define NEG_SLOPE 0.2f

// ---------------- utility kernels ----------------

__global__ void k_fill_f32(float* __restrict__ p, float v, size_t n) {
  size_t i = (size_t)blockIdx.x * blockDim.x + threadIdx.x;
  if (i < n) p[i] = v;
}

__global__ void k_f32_to_f16(const float* __restrict__ in, half_t* __restrict__ out, size_t n) {
  size_t i = (size_t)blockIdx.x * blockDim.x + threadIdx.x;
  if (i < n) out[i] = (half_t)in[i];
}

// Wt[n*K + k] = W[k*N + n]  (store W transposed, N-major, as f16 so that a
// B-fragment column is a single contiguous 32B load)
__global__ void k_transpose_f32_to_f16(const float* __restrict__ W, half_t* __restrict__ Wt,
                                       int K, int N) {
  int i = blockIdx.x * blockDim.x + threadIdx.x;
  if (i >= K * N) return;
  int n = i / K;
  int k = i - n * K;
  Wt[i] = (half_t)W[(size_t)k * N + n];
}

// ---------------- WMMA GEMM: C[M,256] = A[M,256] x W[256,256] ----------------
// A: f16 row-major. Bt: f16, Bt[n*256+k] = W[k][n]. C: f32 row-major.
// Grid: M/16 blocks of 256 threads (8 waves). Wave w computes N-tiles 2w, 2w+1.

__global__ __launch_bounds__(256) void k_gemm_wmma(const half_t* __restrict__ A,
                                                   const half_t* __restrict__ Bt,
                                                   float* __restrict__ C, int M) {
  const int lane    = threadIdx.x & 31;
  const int wave    = threadIdx.x >> 5;
  const int strip   = blockIdx.x;       // 16-row strip of A/C
  const int l16     = lane & 15;
  const int halfSel = lane >> 4;        // 0: lanes 0-15, 1: lanes 16-31

  const int row = strip * 16 + l16;     // M row this lane owns for A
  const int n0  = (2 * wave + 0) * 16;
  const int n1  = (2 * wave + 1) * 16;

  v8f acc0 = {};
  v8f acc1 = {};

  const half_t* __restrict__ arow  = A  + (size_t)row * 256;
  const half_t* __restrict__ bcol0 = Bt + (size_t)(n0 + l16) * 256 + halfSel * 16;
  const half_t* __restrict__ bcol1 = Bt + (size_t)(n1 + l16) * 256 + halfSel * 16;

#pragma unroll
  for (int k0 = 0; k0 < 256; k0 += 32) {
    // A fragment (16x32 f16): elems 0..15 <- K=kb..kb+15 (contiguous, matches B)
    const int kb = k0 + halfSel * 16;
    v8h alo = *(const v8h*)(arow + kb);
    v8h ahi = *(const v8h*)(arow + kb + 8);
    v16h afrag;
#pragma unroll
    for (int i = 0; i < 8; ++i) { afrag[i] = alo[i]; afrag[8 + i] = ahi[i]; }

    // B fragment (32x16 f16): lane holds column n0+l16, 16 contiguous K values
    v16h bf0 = *(const v16h*)(bcol0 + k0);
    v16h bf1 = *(const v16h*)(bcol1 + k0);

    acc0 = __builtin_amdgcn_wmma_f32_16x16x32_f16(false, afrag, false, bf0,
                                                  (short)0, acc0, false, false);
    acc1 = __builtin_amdgcn_wmma_f32_16x16x32_f16(false, afrag, false, bf1,
                                                  (short)0, acc1, false, false);
  }

  // C/D layout: VGPR r -> M = strip*16 + halfSel*8 + r, N = ntile*16 + l16
  const int m0 = strip * 16 + halfSel * 8;
  float* __restrict__ c0 = C + (size_t)m0 * 256 + n0 + l16;
  float* __restrict__ c1 = C + (size_t)m0 * 256 + n1 + l16;
#pragma unroll
  for (int r = 0; r < 8; ++r) {
    c0[(size_t)r * 256] = acc0[r];
    c1[(size_t)r * 256] = acc1[r];
  }
}

// ---------------- per-node attention coefficients ----------------
// One wave per (node, head): as[node*heads+h] = dot(H[node, h*C:(h+1)*C], a_src[h])

__global__ __launch_bounds__(256) void k_alphas(const float* __restrict__ H,
                                                const float* __restrict__ a_src,
                                                const float* __restrict__ a_dst,
                                                float* __restrict__ as_, float* __restrict__ ad_,
                                                int n_nodes, int heads, int Cc) {
  int wave = (int)(((size_t)blockIdx.x * blockDim.x + threadIdx.x) >> 5);
  int lane = threadIdx.x & 31;
  int total = n_nodes * heads;
  if (wave >= total) return;
  int node = wave / heads;
  int head = wave - node * heads;
  const float* __restrict__ hrow = H + (size_t)node * (heads * Cc) + head * Cc;
  const float* __restrict__ sv = a_src + head * Cc;
  const float* __restrict__ dv = a_dst + head * Cc;
  float s = 0.f, d = 0.f;
  for (int c = lane; c < Cc; c += 32) {
    float hv = hrow[c];
    s += hv * sv[c];
    d += hv * dv[c];
  }
#pragma unroll
  for (int off = 16; off > 0; off >>= 1) {
    s += __shfl_xor(s, off, 32);
    d += __shfl_xor(d, off, 32);
  }
  if (lane == 0) { as_[wave] = s; ad_[wave] = d; }
}

// ---------------- segment softmax over destination nodes ----------------

__device__ __forceinline__ void atomic_max_f32(float* addr, float val) {
  // IEEE ordering trick: signed-int max for val>=0, unsigned-int min for val<0.
  if (val >= 0.0f) {
    atomicMax((int*)addr, __float_as_int(val));
  } else {
    atomicMin((unsigned int*)addr, __float_as_uint(val));
  }
}

__device__ __forceinline__ void edge_ends(const int* __restrict__ ei, int e, int E,
                                          int& s, int& d) {
  if (e < E) { s = ei[e]; d = ei[E + e]; }
  else       { s = e - E; d = e - E; }       // self loops appended
}

__global__ void k_edge_max(const int* __restrict__ ei, int E, int n_nodes,
                           const float* __restrict__ as_, const float* __restrict__ ad_,
                           float* __restrict__ emax, int heads) {
  int e = blockIdx.x * blockDim.x + threadIdx.x;
  if (e >= E + n_nodes) return;
  int s, d;
  edge_ends(ei, e, E, s, d);
  for (int h = 0; h < heads; ++h) {
    float v = as_[s * heads + h] + ad_[d * heads + h];
    v = (v > 0.f) ? v : v * NEG_SLOPE;
    atomic_max_f32(&emax[d * heads + h], v);
  }
}

__global__ void k_edge_expsum(const int* __restrict__ ei, int E, int n_nodes,
                              const float* __restrict__ as_, const float* __restrict__ ad_,
                              const float* __restrict__ emax, float* __restrict__ eexp,
                              float* __restrict__ denom, int heads) {
  int e = blockIdx.x * blockDim.x + threadIdx.x;
  if (e >= E + n_nodes) return;
  int s, d;
  edge_ends(ei, e, E, s, d);
  for (int h = 0; h < heads; ++h) {
    float v = as_[s * heads + h] + ad_[d * heads + h];
    v = (v > 0.f) ? v : v * NEG_SLOPE;
    float w = __expf(v - emax[d * heads + h]);
    eexp[(size_t)e * heads + h] = w;
    atomicAdd(&denom[d * heads + h], w);
  }
}

// One wave per edge; lane covers channels lane, lane+32, ... (width = heads*Cc = 256)
__global__ __launch_bounds__(256) void k_edge_aggregate(const int* __restrict__ ei, int E,
                                                        int n_nodes,
                                                        const float* __restrict__ eexp,
                                                        const float* __restrict__ denom,
                                                        const float* __restrict__ H,
                                                        float* __restrict__ out,
                                                        int heads, int Cc) {
  int lane = threadIdx.x & 31;
  int e = blockIdx.x * 8 + (threadIdx.x >> 5);
  if (e >= E + n_nodes) return;
  int s, d;
  edge_ends(ei, e, E, s, d);
  const int width = heads * Cc;  // 256 for both layers
  const float* __restrict__ hs = H + (size_t)s * width;
  float* __restrict__ od = out + (size_t)d * width;
  __builtin_prefetch(hs, 0, 0);  // global_prefetch_b8 for the gathered source row
  for (int c = lane; c < width; c += 32) {
    int h = c / Cc;
    float alpha = eexp[(size_t)e * heads + h] / denom[(size_t)d * heads + h];
    atomicAdd(&od[c], alpha * hs[c]);
  }
}

// ---------------- pointwise: bias(+elu)(+f16 downconvert) ----------------

__global__ void k_elu_bias_to_f16(const float* __restrict__ in, const float* __restrict__ b,
                                  half_t* __restrict__ outh, size_t n, int width) {
  size_t i = (size_t)blockIdx.x * blockDim.x + threadIdx.x;
  if (i >= n) return;
  float v = in[i] + b[i % width];
  v = (v > 0.f) ? v : (__expf(v) - 1.f);  // elu, alpha = 1
  outh[i] = (half_t)v;
}

__global__ void k_bias_add(float* __restrict__ out, const float* __restrict__ b,
                           size_t n, int width) {
  size_t i = (size_t)blockIdx.x * blockDim.x + threadIdx.x;
  if (i >= n) return;
  out[i] += b[i % width];
}

// ---------------------------------------------------------------------------

static inline size_t alignup256(size_t v) { return (v + 255) & ~(size_t)255; }
static inline int cdiv(size_t a, size_t b) { return (int)((a + b - 1) / b); }

extern "C" void kernel_launch(void* const* d_in, const int* in_sizes, int n_in,
                              void* d_out, int out_size, void* d_ws, size_t ws_size,
                              hipStream_t stream) {
  const float* x      = (const float*)d_in[0];
  const int*   ei     = (const int*)  d_in[1];  // harness contract: integer -> const int*
  const float* W1     = (const float*)d_in[2];
  const float* a_src1 = (const float*)d_in[3];
  const float* a_dst1 = (const float*)d_in[4];
  const float* b1     = (const float*)d_in[5];
  const float* W2     = (const float*)d_in[6];
  const float* a_src2 = (const float*)d_in[7];
  const float* a_dst2 = (const float*)d_in[8];
  const float* b2     = (const float*)d_in[9];

  const int n_nodes = in_sizes[0] / 256;      // IN_CH = 256
  const int E       = in_sizes[1] / 2;        // edge_index is (2, E)
  const int ET      = E + n_nodes;            // + self loops
  const size_t nn   = (size_t)n_nodes;
  const size_t feat = nn * 256;

  // ---- workspace carve-up (everything re-initialized every call) ----
  char* ws = (char*)d_ws;
  size_t off = 0;
  half_t* Xh   = (half_t*)(ws + off); off += alignup256(feat * sizeof(half_t));   // X (f16); reused as elu(h1) f16
  half_t* W1h  = (half_t*)(ws + off); off += alignup256(256 * 256 * sizeof(half_t));
  half_t* W2h  = (half_t*)(ws + off); off += alignup256(256 * 256 * sizeof(half_t));
  float*  H    = (float*) (ws + off); off += alignup256(feat * sizeof(float));    // H1; reused as H2
  float*  out1 = (float*) (ws + off); off += alignup256(feat * sizeof(float));
  float*  as_  = (float*) (ws + off); off += alignup256(nn * 4 * sizeof(float));
  float*  ad_  = (float*) (ws + off); off += alignup256(nn * 4 * sizeof(float));
  float*  emax = (float*) (ws + off); off += alignup256(nn * 4 * sizeof(float));
  float*  den  = (float*) (ws + off); off += alignup256(nn * 4 * sizeof(float));
  float*  eexp = (float*) (ws + off); off += alignup256((size_t)ET * 4 * sizeof(float));

  const float NEG_INF = -__builtin_inff();
  const int T = 256;

  // ---- prep: f16 copies ----
  k_f32_to_f16<<<cdiv(feat, T), T, 0, stream>>>(x, Xh, feat);
  k_transpose_f32_to_f16<<<cdiv(256 * 256, T), T, 0, stream>>>(W1, W1h, 256, 256);
  k_transpose_f32_to_f16<<<cdiv(256 * 256, T), T, 0, stream>>>(W2, W2h, 256, 256);

  // ================= layer 1: GATConv(256 -> 4 heads x 64) =================
  k_gemm_wmma<<<n_nodes / 16, T, 0, stream>>>(Xh, W1h, H, n_nodes);  // 50000 = 3125*16
  k_alphas<<<cdiv(nn * 4, 8), T, 0, stream>>>(H, a_src1, a_dst1, as_, ad_, n_nodes, 4, 64);

  k_fill_f32<<<cdiv(nn * 4, T), T, 0, stream>>>(emax, NEG_INF, nn * 4);
  k_fill_f32<<<cdiv(nn * 4, T), T, 0, stream>>>(den, 0.f, nn * 4);
  k_fill_f32<<<cdiv(feat, T), T, 0, stream>>>(out1, 0.f, feat);

  k_edge_max<<<cdiv(ET, T), T, 0, stream>>>(ei, E, n_nodes, as_, ad_, emax, 4);
  k_edge_expsum<<<cdiv(ET, T), T, 0, stream>>>(ei, E, n_nodes, as_, ad_, emax, eexp, den, 4);
  k_edge_aggregate<<<cdiv(ET, 8), T, 0, stream>>>(ei, E, n_nodes, eexp, den, H, out1, 4, 64);

  // elu(out1 + b1) -> f16, reusing Xh as the layer-2 GEMM input
  k_elu_bias_to_f16<<<cdiv(feat, T), T, 0, stream>>>(out1, b1, Xh, feat, 256);

  // ================= layer 2: GATConv(256 -> 1 head x 256) =================
  k_gemm_wmma<<<n_nodes / 16, T, 0, stream>>>(Xh, W2h, H, n_nodes);
  k_alphas<<<cdiv(nn * 1, 8), T, 0, stream>>>(H, a_src2, a_dst2, as_, ad_, n_nodes, 1, 256);

  k_fill_f32<<<cdiv(nn, T), T, 0, stream>>>(emax, NEG_INF, nn);
  k_fill_f32<<<cdiv(nn, T), T, 0, stream>>>(den, 0.f, nn);
  k_fill_f32<<<cdiv(feat, T), T, 0, stream>>>((float*)d_out, 0.f, feat);

  k_edge_max<<<cdiv(ET, T), T, 0, stream>>>(ei, E, n_nodes, as_, ad_, emax, 1);
  k_edge_expsum<<<cdiv(ET, T), T, 0, stream>>>(ei, E, n_nodes, as_, ad_, emax, eexp, den, 1);
  k_edge_aggregate<<<cdiv(ET, 8), T, 0, stream>>>(ei, E, n_nodes, eexp, den, H,
                                                  (float*)d_out, 1, 256);
  k_bias_add<<<cdiv(feat, T), T, 0, stream>>>((float*)d_out, b2, feat, 256);
}
